// TransformerBlock_32427003085219
// MI455X (gfx1250) — compile-verified, matrix operand order checked by reading, not run on hardware
//
#include <hip/hip_runtime.h>
#include <cmath>

typedef __attribute__((ext_vector_type(16))) _Float16 v16h;
typedef __attribute__((ext_vector_type(8)))  _Float16 v8h;
typedef __attribute__((ext_vector_type(8)))  float    v8f;

#define DI __device__ __forceinline__

constexpr int Bn    = 8;
constexpr int C     = 96;
constexpr int D     = 48;      // C / HEADS
constexpr int HID   = 255;
constexpr int HIDP  = 256;     // padded K for final GEMM
constexpr int H     = 128, W = 128;
constexpr int HW    = H * W;   // 16384
constexpr int BH    = 16;      // Bn * HEADS

// ---------------- workspace layout (bytes) ----------------
constexpr size_t MB       = 1ull << 20;
constexpr size_t OFF_W1H  = 0;            // 288*96*2
constexpr size_t OFF_W3H  = 64ull  << 10; // 96*96*2
constexpr size_t OFF_G1H  = 128ull << 10; // 512*96*2
constexpr size_t OFF_G3H  = 256ull << 10; // 96*256*2
constexpr size_t OFF_S    = 512ull << 10; // 16*48*48*4 = 147456
constexpr size_t OFF_ATT  = 704ull << 10; // 16*48*64*2 = 98304
constexpr size_t OFF_NRM  = 960ull << 10; // Bn*192*4 = 6144
constexpr size_t OFF_X1H  = 2   * MB;     // f16 [B][HW][96]   = 24 MiB
constexpr size_t OFF_R1   = 28  * MB;     // 72 MiB: qkv_pre f16 [B][HW][288]; later outh(24M)+mdta f32(48M)
constexpr size_t OFF_R2   = 104 * MB;     // 72 MiB: qkv f16 [B][HW][288]; later x2 f32(48M)+x2h f16(24M)
constexpr size_t OFF_HPRE = 180 * MB;     // f16 [B][HW][512] = 128 MiB; later t f16 [B][HW][256]
constexpr size_t OFF_HB   = 312 * MB;     // f16 [B][HW][512] = 128 MiB   (peak ~440 MiB)

// ---------------- WMMA fragment helpers ----------------
// Contiguous 16-half fragment (16-bit A/B layout): this lane holds K values
// {kb..kb+7} and {kb+16..kb+23}, kb already includes (lane>=16 ? 8 : 0).
DI v16h load_a16(const _Float16* p) {
  v8h lo = *(const v8h*)p;
  v8h hi = *(const v8h*)(p + 16);
  v16h r;
#pragma unroll
  for (int i = 0; i < 8; ++i) { r[i] = lo[i]; r[i + 8] = hi[i]; }
  return r;
}

// Low half only; upper 8 K-values forced to zero (for padded K tails).
DI v16h load_a16_lo(const _Float16* p) {
  v8h lo = *(const v8h*)p;
  v16h r;
#pragma unroll
  for (int i = 0; i < 8; ++i) { r[i] = lo[i]; r[i + 8] = (_Float16)0.f; }
  return r;
}

// Strided fragment: element t<8 at p[t*stride], t>=8 at p[(t+8)*stride] (K-dim strided in memory).
DI v16h load_strided(const _Float16* p, int stride) {
  v16h r;
#pragma unroll
  for (int t = 0; t < 8; ++t) {
    r[t]     = p[(size_t)t * stride];
    r[t + 8] = p[(size_t)(t + 16) * stride];
  }
  return r;
}

DI v8f wmma_f16(v16h a, v16h b, v8f c) {
  return __builtin_amdgcn_wmma_f32_16x16x32_f16(false, a, false, b, (short)0, c, false, false);
}

// ---------------- weight f32 -> f16 (with padding) ----------------
__global__ void conv_weights(const float* __restrict__ w1, const float* __restrict__ w3,
                             const float* __restrict__ g1, const float* __restrict__ g3,
                             _Float16* __restrict__ w1h, _Float16* __restrict__ w3h,
                             _Float16* __restrict__ g1h, _Float16* __restrict__ g3h) {
  int idx = blockIdx.x * blockDim.x + threadIdx.x;   // [0, 512*96)
  if (idx < 288 * 96) w1h[idx] = (_Float16)w1[idx];
  if (idx < 96 * 96)  w3h[idx] = (_Float16)w3[idx];
  if (idx < 512 * 96) {
    int row = idx / 96;
    g1h[idx] = (row < 510) ? (_Float16)g1[idx] : (_Float16)0.f;
  }
  if (idx < 96 * 256) {
    int row = idx / 256, cc = idx % 256;
    g3h[idx] = (cc < HID) ? (_Float16)g3[row * HID + cc] : (_Float16)0.f;
  }
}

// ---------------- channel LayerNorm: planar f32 in -> pixel-major f16 (+opt planar f32) ----------------
template <bool W32>
__global__ void ln_chan(const float* __restrict__ in, float* __restrict__ out32,
                        _Float16* __restrict__ out16,
                        const float* __restrict__ gw, const float* __restrict__ gb) {
  int p = blockIdx.x * blockDim.x + threadIdx.x;     // pixel id in [0, Bn*HW)
  if (p >= Bn * HW) return;
  int b = p >> 14;
  int pix = p & (HW - 1);
  const float* base = in + (size_t)b * C * HW + pix;
  float vals[C];
  float s = 0.f;
#pragma unroll
  for (int c = 0; c < C; ++c) { vals[c] = base[(size_t)c * HW]; s += vals[c]; }
  float mean = s * (1.f / C);
  float v = 0.f;
#pragma unroll
  for (int c = 0; c < C; ++c) { float d = vals[c] - mean; v += d * d; }
  float rstd = rsqrtf(v * (1.f / C) + 1e-5f);
  _Float16 o16[C];
  for (int c = 0; c < C; ++c) {
    float y = (vals[c] - mean) * rstd * gw[c] + gb[c];
    o16[c] = (_Float16)y;
    if constexpr (W32) out32[(size_t)b * C * HW + (size_t)c * HW + pix] = y;
  }
  v8h* dst = (v8h*)(out16 + ((size_t)b * HW + pix) * C);
#pragma unroll
  for (int k = 0; k < C / 8; ++k) {
    v8h t;
#pragma unroll
    for (int i = 0; i < 8; ++i) t[i] = o16[k * 8 + i];
    dst[k] = t;
  }
}

// ---------------- WMMA GEMM: out[b] = A[M][K] x act[b][HW][K]^T ----------------
// act pixel-major (K contiguous per pixel). f16 out: pixel-major [b][HW][KsOut].
// f32 out: planar [b][M][HW] with optional residual.
template <bool OUT_F16, bool RES>
__global__ void gemm_wmma(const _Float16* __restrict__ A, const _Float16* __restrict__ Bact,
                          void* __restrict__ outp, const float* __restrict__ res,
                          int K, int Mtot, int KsOut) {
  int lane  = threadIdx.x;           // one wave per block
  int col   = lane & 15;
  int hi    = lane >> 4;
  int nBase = blockIdx.x * 64;
  int mBase = blockIdx.y * 16;
  int b     = blockIdx.z;
  const _Float16* Bb = Bact + (size_t)b * HW * K;
  v8f acc[4] = {};
  for (int kt = 0; kt < K; kt += 32) {
    int kk = kt + hi * 8;
    v16h a = load_a16(A + (size_t)(mBase + col) * K + kk);
#pragma unroll
    for (int j = 0; j < 4; ++j) {
      int n = nBase + j * 16 + col;
      v16h bt = load_a16(Bb + (size_t)n * K + kk);     // contiguous fragment
      acc[j] = wmma_f16(a, bt, acc[j]);
    }
  }
#pragma unroll
  for (int j = 0; j < 4; ++j) {
    int n = nBase + j * 16 + col;
    if constexpr (OUT_F16) {
      v8h hv;
#pragma unroll
      for (int r = 0; r < 8; ++r) hv[r] = (_Float16)acc[j][r];
      *(v8h*)((_Float16*)outp + ((size_t)b * HW + n) * KsOut + mBase + hi * 8) = hv;
    } else {
#pragma unroll
      for (int r = 0; r < 8; ++r) {
        int row = mBase + hi * 8 + r;
        size_t o = (size_t)b * Mtot * HW + (size_t)row * HW + n;
        float v = acc[j][r];
        if constexpr (RES) v += res[o];
        ((float*)outp)[o] = v;
      }
    }
  }
}

// ---------------- grouped 3x3 depthwise conv, pixel-major f16 ----------------
__global__ void dwconv3x3(const _Float16* __restrict__ in, _Float16* __restrict__ out,
                          const float* __restrict__ w, int Cout, int cpgIn, int cpgOut,
                          int inS, int outS) {
  size_t idx   = (size_t)blockIdx.x * blockDim.x + threadIdx.x;
  size_t total = (size_t)Bn * HW * Cout;
  if (idx >= total) return;
  int o = (int)(idx % Cout);            // channel fastest -> coalesced
  size_t rest = idx / Cout;
  int pix = (int)(rest & (HW - 1));
  int b = (int)(rest >> 14);
  int x = pix & (W - 1);
  int y = pix >> 7;
  int g  = o / cpgOut;
  int ib = g * cpgIn;
  const _Float16* inb = in + (size_t)b * HW * inS;
  float acc = 0.f;
  for (int i = 0; i < cpgIn; ++i) {
    const float* wp = w + ((size_t)o * cpgIn + i) * 9;
    int ch = ib + i;
#pragma unroll
    for (int ky = -1; ky <= 1; ++ky) {
      int yy = y + ky;
      if (yy < 0 || yy >= H) continue;
#pragma unroll
      for (int kx = -1; kx <= 1; ++kx) {
        int xx = x + kx;
        if (xx < 0 || xx >= W) continue;
        acc += wp[(ky + 1) * 3 + (kx + 1)] * (float)inb[(size_t)(yy * W + xx) * inS + ch];
      }
    }
  }
  out[((size_t)b * HW + pix) * outS + o] = (_Float16)acc;
}

// ---------------- per-channel sum of squares for q,k (pixel-major, coalesced) ----------------
__global__ void norm_partial(const _Float16* __restrict__ qkv, float* __restrict__ sums) {
  int c = threadIdx.x;                  // 0..191 (q: 0..95, k: 96..191)
  int chunk = blockIdx.x;               // Bn * (HW/256) blocks
  int b = chunk >> 6;
  int p0 = (chunk & 63) * 256;
  const _Float16* base = qkv + ((size_t)b * HW + p0) * 288 + c;
  float s = 0.f;
  for (int p = 0; p < 256; ++p) { float v = (float)base[(size_t)p * 288]; s += v * v; }
  atomicAdd(&sums[b * 192 + c], s);
}

__global__ void norm_finalize(float* __restrict__ sums) {
  int i = blockIdx.x * blockDim.x + threadIdx.x;
  if (i < Bn * 192) sums[i] = 1.f / fmaxf(sqrtf(sums[i]), 1e-12f);
}

// ---------------- S += q x k^T over a K(=pixel) chunk, raw (norms folded into softmax) ----------------
__global__ void attn_qk(const _Float16* __restrict__ qkv, float* __restrict__ S) {
  int lane = threadIdx.x;
  int col = lane & 15, hi = lane >> 4;
  int bh = blockIdx.y;
  int b = bh >> 1, head = bh & 1;
  const _Float16* qbase = qkv + (size_t)b * HW * 288 + head * D;
  const _Float16* kbase = qkv + (size_t)b * HW * 288 + C + head * D;
  int k0 = blockIdx.x * (HW / 8);       // 8 K-chunks of 2048 pixels
  v8f acc[3][3] = {};
  for (int kt = k0; kt < k0 + HW / 8; kt += 32) {
    int kk = kt + hi * 8;
    v16h a[3], bt[3];
#pragma unroll
    for (int i = 0; i < 3; ++i) a[i]  = load_strided(qbase + (size_t)kk * 288 + i * 16 + col, 288);
#pragma unroll
    for (int i = 0; i < 3; ++i) bt[i] = load_strided(kbase + (size_t)kk * 288 + i * 16 + col, 288);
#pragma unroll
    for (int mi = 0; mi < 3; ++mi)
#pragma unroll
      for (int ni = 0; ni < 3; ++ni)
        acc[mi][ni] = wmma_f16(a[mi], bt[ni], acc[mi][ni]);
  }
#pragma unroll
  for (int mi = 0; mi < 3; ++mi)
#pragma unroll
    for (int ni = 0; ni < 3; ++ni)
#pragma unroll
      for (int r = 0; r < 8; ++r) {
        int row = mi * 16 + hi * 8 + r;
        int cc  = ni * 16 + col;
        atomicAdd(&S[((size_t)bh * D + row) * D + cc], acc[mi][ni][r]);
      }
}

// ---------------- softmax with folded l2 norms + scaling -> f16 attn padded to K=64 ----------------
__global__ void softmax_rows(const float* __restrict__ S, const float* __restrict__ scal,
                             const float* __restrict__ nrm, _Float16* __restrict__ attnh) {
  int r = blockIdx.x * blockDim.x + threadIdx.x;
  if (r >= BH * D) return;
  int bh = r / D, m = r % D;
  int b = bh >> 1, head = bh & 1;
  float sc = scal[head] * nrm[b * 192 + head * D + m];      // * inv||q_m||
  const float* row = S + ((size_t)bh * D + m) * D;
  const float* invk = nrm + b * 192 + C + head * D;         // inv||k_n||
  float vals[D];
  float mx = -3.4e38f;
  for (int j = 0; j < D; ++j) { vals[j] = row[j] * sc * invk[j]; mx = fmaxf(mx, vals[j]); }
  float sum = 0.f;
  for (int j = 0; j < D; ++j) { vals[j] = __expf(vals[j] - mx); sum += vals[j]; }
  float inv = 1.f / sum;
  _Float16* o = attnh + ((size_t)bh * D + m) * 64;
  for (int j = 0; j < D; ++j) o[j] = (_Float16)(vals[j] * inv);
  for (int j = D; j < 64; ++j) o[j] = (_Float16)0.f;
}

// ---------------- out = attn x v (M=48, K=48 pad 64, N=HW); v pixel-major => contiguous B ----------------
__global__ void attn_v(const _Float16* __restrict__ attnh, const _Float16* __restrict__ qkv,
                       _Float16* __restrict__ outh) {
  int lane = threadIdx.x;
  int col = lane & 15, hi = lane >> 4;
  int nBase = blockIdx.x * 64;
  int mBase = blockIdx.y * 16;
  int bh = blockIdx.z;
  int b = bh >> 1, head = bh & 1;
  const _Float16* A  = attnh + (size_t)bh * D * 64;
  const _Float16* vb = qkv + (size_t)b * HW * 288 + 2 * C + head * D;
  v8f acc[4] = {};
  // k-step 0: K 0..31, all valid
  {
    int kk = hi * 8;
    v16h a = load_a16(A + (size_t)(mBase + col) * 64 + kk);
#pragma unroll
    for (int j = 0; j < 4; ++j) {
      int n = nBase + j * 16 + col;
      v16h bt = load_a16(vb + (size_t)n * 288 + kk);
      acc[j] = wmma_f16(a, bt, acc[j]);
    }
  }
  // k-step 1: K 32..63; upper half (48..63) zero (attn cols are zero-padded too)
  {
    int kk = 32 + hi * 8;
    v16h a = load_a16(A + (size_t)(mBase + col) * 64 + kk);
#pragma unroll
    for (int j = 0; j < 4; ++j) {
      int n = nBase + j * 16 + col;
      v16h bt = load_a16_lo(vb + (size_t)n * 288 + kk);
      acc[j] = wmma_f16(a, bt, acc[j]);
    }
  }
#pragma unroll
  for (int j = 0; j < 4; ++j) {
    int n = nBase + j * 16 + col;
    v8h hv;
#pragma unroll
    for (int r = 0; r < 8; ++r) hv[r] = (_Float16)acc[j][r];
    *(v8h*)(outh + ((size_t)b * HW + n) * C + head * D + mBase + hi * 8) = hv;
  }
}

// ---------------- gated GELU: t[ch] = gelu(h[255+ch]) * h[ch], pad ch==255; pixel-major ----------------
__global__ void gated_gelu(const _Float16* __restrict__ h, _Float16* __restrict__ t) {
  size_t idx = (size_t)blockIdx.x * blockDim.x + threadIdx.x;
  if (idx >= (size_t)Bn * HW * HIDP) return;
  int ch = (int)(idx & 255);            // channel fastest -> coalesced
  size_t rest = idx >> 8;
  int pix = (int)(rest & (HW - 1));
  int b = (int)(rest >> 14);
  _Float16 r;
  if (ch >= HID) {
    r = (_Float16)0.f;
  } else {
    const _Float16* hp = h + ((size_t)b * HW + pix) * 512;
    float g1 = (float)hp[ch];
    float g2 = (float)hp[HID + ch];
    float ge = 0.5f * g2 * (1.f + erff(g2 * 0.70710678f));
    r = (_Float16)(ge * g1);
  }
  t[((size_t)b * HW + pix) * HIDP + ch] = r;
}

// ---------------- host orchestration ----------------
extern "C" void kernel_launch(void* const* d_in, const int* in_sizes, int n_in,
                              void* d_out, int out_size, void* d_ws, size_t ws_size,
                              hipStream_t stream) {
  (void)in_sizes; (void)n_in; (void)out_size; (void)ws_size;
  const float* x    = (const float*)d_in[0];
  const float* lnw  = (const float*)d_in[1];
  const float* lnb  = (const float*)d_in[2];
  const float* scal = (const float*)d_in[3];
  const float* w1   = (const float*)d_in[4];   // (288, 96)
  const float* w2   = (const float*)d_in[5];   // (288, 1, 3, 3)
  const float* w3   = (const float*)d_in[6];   // (96, 96)
  const float* gw1  = (const float*)d_in[7];   // (510, 96)
  const float* gw2  = (const float*)d_in[8];   // (510, 2, 3, 3)
  const float* gw3  = (const float*)d_in[9];   // (96, 255)
  float* out = (float*)d_out;
  char* ws = (char*)d_ws;

  _Float16* w1h  = (_Float16*)(ws + OFF_W1H);
  _Float16* w3h  = (_Float16*)(ws + OFF_W3H);
  _Float16* g1h  = (_Float16*)(ws + OFF_G1H);
  _Float16* g3h  = (_Float16*)(ws + OFF_G3H);
  float*    S    = (float*)   (ws + OFF_S);
  _Float16* att  = (_Float16*)(ws + OFF_ATT);
  float*    nrm  = (float*)   (ws + OFF_NRM);
  _Float16* x1h  = (_Float16*)(ws + OFF_X1H);              // [B][HW][96]
  _Float16* qkvp = (_Float16*)(ws + OFF_R1);               // [B][HW][288]
  _Float16* qkv  = (_Float16*)(ws + OFF_R2);               // [B][HW][288]
  _Float16* outh = (_Float16*)(ws + OFF_R1);               // reuse R1: [B][HW][96]
  float*    mdta = (float*)   (ws + OFF_R1 + 24 * MB);     // planar f32 [B][96][HW]
  float*    x2   = (float*)   (ws + OFF_R2);               // reuse R2: planar f32
  _Float16* x2h  = (_Float16*)(ws + OFF_R2 + 48 * MB);     // [B][HW][96]
  _Float16* hpre = (_Float16*)(ws + OFF_HPRE);             // [B][HW][512]
  _Float16* hbuf = (_Float16*)(ws + OFF_HB);               // [B][HW][512]
  _Float16* tbuf = (_Float16*)(ws + OFF_HPRE);             // reuse: [B][HW][256]

  // 0) weights -> f16 (padded)
  conv_weights<<<192, 256, 0, stream>>>(w1, w3, gw1, gw3, w1h, w3h, g1h, g3h);

  // 1) x1 = LN(x) -> pixel-major f16
  ln_chan<false><<<(Bn * HW) / 128, 128, 0, stream>>>(x, nullptr, x1h, lnw, lnb);

  // 2) qkv_pre = W1 x x1  (M=288, K=96) -> pixel-major f16
  gemm_wmma<true, false><<<dim3(HW / 64, 288 / 16, Bn), 32, 0, stream>>>(
      w1h, x1h, qkvp, nullptr, 96, 288, 288);

  // 3) depthwise 3x3, groups=288
  dwconv3x3<<<(unsigned)(((size_t)Bn * HW * 288 + 255) / 256), 256, 0, stream>>>(
      qkvp, qkv, w2, 288, 1, 1, 288, 288);

  // 4) per-channel inverse l2 norms of q,k (folded into softmax later)
  hipMemsetAsync(nrm, 0, (size_t)Bn * 192 * sizeof(float), stream);
  norm_partial<<<Bn * (HW / 256), 192, 0, stream>>>(qkv, nrm);
  norm_finalize<<<(Bn * 192 + 255) / 256, 256, 0, stream>>>(nrm);

  // 5) S = q x k^T raw (split-K over pixels, atomics)
  hipMemsetAsync(S, 0, (size_t)BH * D * D * sizeof(float), stream);
  attn_qk<<<dim3(8, BH), 32, 0, stream>>>(qkv, S);

  // 6) softmax(scale * inv||q|| * inv||k|| * S) -> f16 attn (padded to K=64)
  softmax_rows<<<(BH * D + 127) / 128, 128, 0, stream>>>(S, scal, nrm, att);

  // 7) out = attn x v  (M=48, N=HW) -> pixel-major f16
  attn_v<<<dim3(HW / 64, 3, BH), 32, 0, stream>>>(att, qkv, outh);

  // 8) mdta_out = W3 x out  (M=96, K=96), planar f32
  gemm_wmma<false, false><<<dim3(HW / 64, 6, Bn), 32, 0, stream>>>(
      w3h, outh, mdta, nullptr, 96, 96, 0);

  // 9) x2 = LN(mdta_out) -> planar f32 + pixel-major f16
  ln_chan<true><<<(Bn * HW) / 128, 128, 0, stream>>>(mdta, x2, x2h, lnw, lnb);

  // 10) h_pre = GW1 x x2  (M=512 padded, K=96) -> pixel-major f16
  gemm_wmma<true, false><<<dim3(HW / 64, 32, Bn), 32, 0, stream>>>(
      g1h, x2h, hpre, nullptr, 96, 512, 512);

  // 11) grouped depthwise 3x3 (groups=255, 2 in / 2 out per group)
  dwconv3x3<<<(unsigned)(((size_t)Bn * HW * 510 + 255) / 256), 256, 0, stream>>>(
      hpre, hbuf, gw2, 510, 2, 2, 512, 512);

  // 12) t = gelu(g2) * g1, padded to 256 channels -> pixel-major f16
  gated_gelu<<<(unsigned)(((size_t)Bn * HW * HIDP) / 256), 256, 0, stream>>>(hbuf, tbuf);

  // 13) y = GW3 x t + x2  (M=96, K=256), planar f32 out with residual
  gemm_wmma<false, true><<<dim3(HW / 64, 6, Bn), 32, 0, stream>>>(
      g3h, tbuf, (void*)out, x2, 256, 96, 0);
}
